// MambaCore_9440338116828
// MI455X (gfx1250) — matrix operand order repaired and checked by reading, not compiled, not run on hardware
//
#include <hip/hip_runtime.h>
#include <math.h>

// ---------------------------------------------------------------------------
// Mamba forward (2 layers) for gfx1250 (MI455X).
// GEMMs: TDM (tensor_load_to_lds) stages A/W tiles into LDS, 8 waves consume
//        via ds_load -> v_wmma_f32_16x16x32_bf16 (bf16 in, f32 accumulate).
// Scan:  fp32 register recurrence, 16 states/thread, v_exp_f32.
// ---------------------------------------------------------------------------

#define D_MODEL   768
#define N_LAYERS  2
#define D_STATE   16
#define D_CONV    4
#define D_INNER   1536
#define DT_RANK   48
#define DT_PAD    64
#define T_LEN     1024
#define B_SZ      4
#define M_ROWS    (T_LEN * B_SZ)          // 4096 tokens
#define XP_OUT    (DT_RANK + 2 * D_STATE) // 80

typedef __attribute__((ext_vector_type(16))) __bf16 v16bf;
typedef __attribute__((ext_vector_type(8)))  float  v8f;
typedef __attribute__((ext_vector_type(4)))  unsigned v4u;
typedef __attribute__((ext_vector_type(8)))  int      v8i_;
typedef __attribute__((ext_vector_type(4)))  int      v4i_;

struct __align__(16) U4 { unsigned x, y, z, w; };

// float -> bf16 (RNE), stored as raw u16 so we never need __bf16 scalar math
__device__ __forceinline__ unsigned short f2bf(float f) {
    unsigned u = __float_as_uint(f);
    unsigned r = (u + 0x7FFFu + ((u >> 16) & 1u)) >> 16;
    return (unsigned short)r;
}

__device__ __forceinline__ float sigmoidf_(float v) { return 1.0f / (1.0f + __expf(-v)); }

// ---------------------------------------------------------------------------
// elementwise converters
// ---------------------------------------------------------------------------
__global__ __launch_bounds__(256) void cvt_f32_bf16(const float* __restrict__ s,
                                                    unsigned short* __restrict__ d, int n) {
    int i = blockIdx.x * 256 + threadIdx.x;
    if (i < n) d[i] = f2bf(s[i]);
}

// pad/convert: dst[r, c] = c < srcCols ? bf16(src[r*srcStride + c]) : 0
__global__ __launch_bounds__(256) void cvt_pad_bf16(const float* __restrict__ s, int srcStride,
                                                    int srcCols, unsigned short* __restrict__ d,
                                                    int dstCols, int rows) {
    int i = blockIdx.x * 256 + threadIdx.x;
    if (i >= rows * dstCols) return;
    int r = i / dstCols, c = i - r * dstCols;
    d[i] = (c < srcCols) ? f2bf(s[(size_t)r * srcStride + c]) : (unsigned short)0;
}

// ---------------------------------------------------------------------------
// Tensor Data Mover: 2D tile (tile_rows x tile_cols, 2-byte elems) -> LDS.
// D# per ISA 8.3/8.4: group0 = count/lds_addr/global_addr/type,
// group1 = data_size, tensor dims (OOB zero-fill), tile dims, dim0 stride.
// 6-arg builtin (clang-23 lane): (v4u, v8i, v4i, v4i, v8i, i32 cpol).
// ---------------------------------------------------------------------------
__device__ __forceinline__ void tdm_load_2d(unsigned lds_off, const unsigned short* gptr,
                                            int rowsRemain, int colsRemain,
                                            int tile_rows, int tile_cols, int stride_elems) {
    unsigned long long ga = (unsigned long long)(size_t)gptr;
    v4u g0;
    g0[0] = 1u;                                   // count=1, user descriptor
    g0[1] = lds_off;                              // LDS byte address
    g0[2] = (unsigned)(ga & 0xFFFFFFFFu);         // global_addr[31:0]
    g0[3] = (unsigned)((ga >> 32) & 0x1FFFFFFu)   // global_addr[56:32]
            | (2u << 30);                         // type = 2 ("image")
    unsigned d0 = (unsigned)colsRemain;           // tensor_dim0 (K units left)
    unsigned d1 = (unsigned)rowsRemain;           // tensor_dim1 (rows left)
    v8i_ g1;
    g1[0] = (int)(1u << 16);                                      // data_size = 2B
    g1[1] = (int)((d0 & 0xFFFFu) << 16);                          // tensor_dim0[15:0]
    g1[2] = (int)((d0 >> 16) | ((d1 & 0xFFFFu) << 16));           // dim0[31:16] | dim1[15:0]
    g1[3] = (int)((d1 >> 16) | ((unsigned)tile_cols << 16));      // dim1[31:16] | tile_dim0
    g1[4] = (int)(unsigned)tile_rows;                             // tile_dim1
    g1[5] = stride_elems;                                         // tensor_dim0_stride[31:0]
    g1[6] = 0;
    g1[7] = 0;
    v4i_ z4 = {0, 0, 0, 0};
    v8i_ z8 = {0, 0, 0, 0, 0, 0, 0, 0};
    __builtin_amdgcn_tensor_load_to_lds(g0, g1, z4, z4, z8, 0);
}

// ---------------------------------------------------------------------------
// WMMA fragment loaders (contiguous k order, wave32) — from LDS tiles
// ---------------------------------------------------------------------------
// A (16x32 MxK): lanes 0-15 -> M=lane; lanes 16-31 -> M=lane-16.
// VGPR v (pair e=2v,2v+1): K = half*16 + 2v (contiguous, matches B)
__device__ __forceinline__ v16bf load_a_frag(const unsigned short* __restrict__ A,
                                             int m, int lda, int k0, int lane) {
    const int kh = ((lane >> 4) & 1) * 16;
    union { v16bf v; unsigned u[8]; } f;
    const unsigned short* row = A + (size_t)m * lda;
#pragma unroll
    for (int r = 0; r < 8; ++r) {
        int k = k0 + kh + 2 * r;
        f.u[r] = *reinterpret_cast<const unsigned*>(row + k);
    }
    return f.v;
}

// B (32x16 KxN): lane L -> N = L&15 ; elements e = B[kblock+e, N], kblock = (L>=16)*16.
// With W stored (N x K) row-major this is 16 contiguous bf16 = two 16B loads.
__device__ __forceinline__ v16bf load_b_frag(const unsigned short* __restrict__ W,
                                             int n, int ldb, int k0, int lane) {
    const int kb = k0 + ((lane >> 4) << 4);
    union { v16bf v; U4 q[2]; } f;
    const U4* p = reinterpret_cast<const U4*>(W + (size_t)n * ldb + kb);
    f.q[0] = p[0];
    f.q[1] = p[1];
    return f.v;
}

// ---------------------------------------------------------------------------
// C = A(MxK,bf16) * W(NxK,bf16)^T  -> f32, NT form.
// Block: 256 thr = 8 waves; wave = 32x32 tile (2x2 WMMA); block tile 64x128.
// Per K-chunk (CK): wave0 issues two TDM DMAs (A 64xCK, W 128xCK) -> LDS,
// s_wait_tensorcnt, barrier, then all waves run CK/32 WMMA steps from LDS.
// TDM OOB zero-fill handles the N=80 partial tile for free.
// EPI: 0 = store, 1 = softplus(acc + bias[col])
// ---------------------------------------------------------------------------
template <int EPI>
__global__ __launch_bounds__(256) void gemm_bf16_nt(const unsigned short* __restrict__ A,
                                                    const unsigned short* __restrict__ W,
                                                    float* __restrict__ C, int M, int N, int K,
                                                    int CK, const float* __restrict__ bias) {
    extern __shared__ unsigned short smem[];
    unsigned short* sA = smem;             // 64  x CK
    unsigned short* sW = smem + 64 * CK;   // 128 x CK

    const int lane = threadIdx.x & 31;
    const int wid  = threadIdx.x >> 5;
    const int wr = wid >> 2, wc = wid & 3;
    const int blockRow = blockIdx.y * 64;
    const int blockCol = blockIdx.x * 128;
    const int mL = lane & 15;

    v8f acc00 = {}, acc01 = {}, acc10 = {}, acc11 = {};

    for (int k0 = 0; k0 < K; k0 += CK) {
        if (wid == 0) {
            tdm_load_2d(0u, A + (size_t)blockRow * K + k0,
                        M - blockRow, K - k0, 64, CK, K);
            tdm_load_2d((unsigned)(64 * CK * 2), W + (size_t)blockCol * K + k0,
                        N - blockCol, K - k0, 128, CK, K);
            __builtin_amdgcn_s_wait_tensorcnt(0);
        }
        __syncthreads();

        for (int kc = 0; kc < CK; kc += 32) {
            v16bf a0 = load_a_frag(sA, wr * 32 + mL,      CK, kc, lane);
            v16bf a1 = load_a_frag(sA, wr * 32 + 16 + mL, CK, kc, lane);
            v16bf b0 = load_b_frag(sW, wc * 32 + mL,      CK, kc, lane);
            v16bf b1 = load_b_frag(sW, wc * 32 + 16 + mL, CK, kc, lane);
            acc00 = __builtin_amdgcn_wmma_f32_16x16x32_bf16(false, a0, false, b0, (short)0, acc00, false, false);
            acc01 = __builtin_amdgcn_wmma_f32_16x16x32_bf16(false, a0, false, b1, (short)0, acc01, false, false);
            acc10 = __builtin_amdgcn_wmma_f32_16x16x32_bf16(false, a1, false, b0, (short)0, acc10, false, false);
            acc11 = __builtin_amdgcn_wmma_f32_16x16x32_bf16(false, a1, false, b1, (short)0, acc11, false, false);
        }
        __syncthreads();
    }

    // C/D layout: VGPR r -> M = r (lanes 0-15) | 8+r (lanes 16-31); N = lane&15
    const int rowBase = blockRow + wr * 32;
    const int colBase = blockCol + wc * 32;
    const int mOff = ((lane >> 4) & 1) * 8;
#pragma unroll
    for (int tm = 0; tm < 2; ++tm) {
#pragma unroll
        for (int tn = 0; tn < 2; ++tn) {
            int col = colBase + tn * 16 + mL;
            if (col >= N) continue;
            v8f acc = (tm == 0) ? (tn == 0 ? acc00 : acc01) : (tn == 0 ? acc10 : acc11);
#pragma unroll
            for (int r = 0; r < 8; ++r) {
                int row = rowBase + tm * 16 + mOff + r;
                float v = acc[r];
                if (EPI == 1) {
                    v += bias[col];
                    v = (v > 20.0f) ? v : log1pf(__expf(v));
                }
                C[(size_t)row * N + col] = v;
            }
        }
    }
}

// ---------------------------------------------------------------------------
// causal depthwise conv (width 4) over time + bias + SiLU
// xz rows are (t*B + b); xs half = cols [0,1536)
// ---------------------------------------------------------------------------
__global__ __launch_bounds__(256) void conv_silu(const float* __restrict__ xz,
                                                 const float* __restrict__ wconv,
                                                 const float* __restrict__ bconv,
                                                 float* __restrict__ xs,
                                                 unsigned short* __restrict__ xs_b) {
    int i = blockIdx.x * 256 + threadIdx.x;
    if (i >= M_ROWS * D_INNER) return;
    int row = i / D_INNER, c = i - row * D_INNER;
    int t = row >> 2, b = row & 3;
    float acc = bconv[c];
#pragma unroll
    for (int j = 0; j < D_CONV; ++j) {
        int tt = t - (D_CONV - 1) + j;
        if (tt >= 0)
            acc += wconv[c * D_CONV + j] * xz[(size_t)(tt * B_SZ + b) * (2 * D_INNER) + c];
    }
    float s = acc * sigmoidf_(acc);
    xs[i]   = s;
    xs_b[i] = f2bf(s);
}

// ---------------------------------------------------------------------------
// SSM scan: one thread per (b, d); 16 states in registers; 1024 serial steps
// ---------------------------------------------------------------------------
__global__ __launch_bounds__(256) void ssm_scan(const float* __restrict__ delta,
                                                const float* __restrict__ xs,
                                                const float* __restrict__ xdbl,
                                                const float* __restrict__ A_log,
                                                float* __restrict__ y) {
    int id = blockIdx.x * 256 + threadIdx.x;
    if (id >= B_SZ * D_INNER) return;
    int b = id / D_INNER, d = id - b * D_INNER;

    float a[D_STATE];
#pragma unroll
    for (int n = 0; n < D_STATE; ++n) a[n] = -__expf(A_log[d * D_STATE + n]);
    float h[D_STATE];
#pragma unroll
    for (int n = 0; n < D_STATE; ++n) h[n] = 0.0f;

    for (int t = 0; t < T_LEN; ++t) {
        int row = t * B_SZ + b;
        float dlt = delta[(size_t)row * D_INNER + d];
        float xv  = xs[(size_t)row * D_INNER + d];
        const float4* Bp = reinterpret_cast<const float4*>(xdbl + (size_t)row * XP_OUT + DT_RANK);
        const float4* Cp = reinterpret_cast<const float4*>(xdbl + (size_t)row * XP_OUT + DT_RANK + D_STATE);
        float4 B4[4] = {Bp[0], Bp[1], Bp[2], Bp[3]};
        float4 C4[4] = {Cp[0], Cp[1], Cp[2], Cp[3]};
        const float* Bv = reinterpret_cast<const float*>(B4);
        const float* Cv = reinterpret_cast<const float*>(C4);
        float dux = dlt * xv;
        float yy = 0.0f;
#pragma unroll
        for (int n = 0; n < D_STATE; ++n) {
            h[n] = h[n] * __expf(dlt * a[n]) + dux * Bv[n];
            yy += h[n] * Cv[n];
        }
        y[(size_t)row * D_INNER + d] = yy;
    }
}

// ---------------------------------------------------------------------------
// gating: out = (y + xs*D) * silu(z);  z = xz cols [1536, 3072)
// ---------------------------------------------------------------------------
__global__ __launch_bounds__(256) void gate_silu(const float* __restrict__ yssm,
                                                 const float* __restrict__ xs,
                                                 const float* __restrict__ xz,
                                                 const float* __restrict__ Dp,
                                                 unsigned short* __restrict__ y_b) {
    int i = blockIdx.x * 256 + threadIdx.x;
    if (i >= M_ROWS * D_INNER) return;
    int row = i / D_INNER, c = i - row * D_INNER;
    float z = xz[(size_t)row * (2 * D_INNER) + D_INNER + c];
    float g = (yssm[i] + xs[i] * Dp[c]) * (z * sigmoidf_(z));
    y_b[i] = f2bf(g);
}

// ---------------------------------------------------------------------------
// LayerNorm over last dim (768); one block (256 thr) per row, 3 cols/thread
// ---------------------------------------------------------------------------
__global__ __launch_bounds__(256) void layernorm_row(const float* __restrict__ h,
                                                     const float* __restrict__ g,
                                                     const float* __restrict__ be,
                                                     float* __restrict__ out) {
    __shared__ float s1[256], s2[256];
    int row = blockIdx.x, tid = threadIdx.x;
    const float* hp = h + (size_t)row * D_MODEL;
    float v0 = hp[tid], v1 = hp[tid + 256], v2 = hp[tid + 512];
    s1[tid] = v0 + v1 + v2;
    s2[tid] = v0 * v0 + v1 * v1 + v2 * v2;
    __syncthreads();
    for (int s = 128; s > 0; s >>= 1) {
        if (tid < s) { s1[tid] += s1[tid + s]; s2[tid] += s2[tid + s]; }
        __syncthreads();
    }
    float mean = s1[0] * (1.0f / D_MODEL);
    float var  = s2[0] * (1.0f / D_MODEL) - mean * mean;
    float rstd = rsqrtf(var + 1e-5f);
    float* op = out + (size_t)row * D_MODEL;
    op[tid]       = (v0 - mean) * rstd * g[tid]       + be[tid];
    op[tid + 256] = (v1 - mean) * rstd * g[tid + 256] + be[tid + 256];
    op[tid + 512] = (v2 - mean) * rstd * g[tid + 512] + be[tid + 512];
}

// ---------------------------------------------------------------------------
// host-side orchestration
// ---------------------------------------------------------------------------
extern "C" void kernel_launch(void* const* d_in, const int* in_sizes, int n_in,
                              void* d_out, int out_size, void* d_ws, size_t ws_size,
                              hipStream_t stream) {
    (void)in_sizes; (void)n_in; (void)out_size; (void)ws_size;
    const float* x          = (const float*)d_in[0];
    const float* in_proj_w  = (const float*)d_in[1];
    const float* conv_w     = (const float*)d_in[2];
    const float* conv_b     = (const float*)d_in[3];
    const float* x_proj_w   = (const float*)d_in[4];
    const float* dt_proj_w  = (const float*)d_in[5];
    const float* dt_proj_b  = (const float*)d_in[6];
    const float* A_log      = (const float*)d_in[7];
    const float* Dvec       = (const float*)d_in[8];
    const float* out_proj_w = (const float*)d_in[9];
    const float* ln_g       = (const float*)d_in[10];
    const float* ln_b       = (const float*)d_in[11];

    char* p = (char*)d_ws;
    size_t off = 0;
    auto take = [&](size_t bytes) { size_t r = off; off += (bytes + 255) & ~(size_t)255; return r; };

    unsigned short* w_in_b  = (unsigned short*)(p + take((size_t)2 * D_INNER * D_MODEL * 2));
    unsigned short* w_xp_b  = (unsigned short*)(p + take((size_t)XP_OUT * D_INNER * 2));
    unsigned short* w_dt_b  = (unsigned short*)(p + take((size_t)D_INNER * DT_PAD * 2));
    unsigned short* w_out_b = (unsigned short*)(p + take((size_t)D_MODEL * D_INNER * 2));
    unsigned short* act_b   = (unsigned short*)(p + take((size_t)M_ROWS * D_MODEL * 2));
    float*          xz      = (float*)(p + take((size_t)M_ROWS * 2 * D_INNER * 4));
    float*          xs      = (float*)(p + take((size_t)M_ROWS * D_INNER * 4));
    unsigned short* xs_b    = (unsigned short*)(p + take((size_t)M_ROWS * D_INNER * 2)); // reused as y_b
    float*          xdbl    = (float*)(p + take((size_t)M_ROWS * XP_OUT * 4));
    unsigned short* dt_b    = (unsigned short*)(p + take((size_t)M_ROWS * DT_PAD * 2));
    float*          delta   = (float*)(p + take((size_t)M_ROWS * D_INNER * 4));
    float*          yssm    = (float*)(p + take((size_t)M_ROWS * D_INNER * 4));
    float*          hout    = (float*)(p + take((size_t)M_ROWS * D_MODEL * 4));

    auto cdiv = [](int a, int b) { return (a + b - 1) / b; };
    auto smem_bytes = [](int ck) { return (size_t)(64 + 128) * ck * sizeof(unsigned short); };
    const int EW = M_ROWS * D_INNER;  // elementwise count

    for (int layer = 0; layer < N_LAYERS; ++layer) {
        const float* src_act = (layer == 0) ? x : hout;
        cvt_f32_bf16<<<cdiv(M_ROWS * D_MODEL, 256), 256, 0, stream>>>(src_act, act_b, M_ROWS * D_MODEL);

        cvt_f32_bf16<<<cdiv(2 * D_INNER * D_MODEL, 256), 256, 0, stream>>>(
            in_proj_w + (size_t)layer * 2 * D_INNER * D_MODEL, w_in_b, 2 * D_INNER * D_MODEL);
        cvt_f32_bf16<<<cdiv(XP_OUT * D_INNER, 256), 256, 0, stream>>>(
            x_proj_w + (size_t)layer * XP_OUT * D_INNER, w_xp_b, XP_OUT * D_INNER);
        cvt_pad_bf16<<<cdiv(D_INNER * DT_PAD, 256), 256, 0, stream>>>(
            dt_proj_w + (size_t)layer * D_INNER * DT_RANK, DT_RANK, DT_RANK, w_dt_b, DT_PAD, D_INNER);
        cvt_f32_bf16<<<cdiv(D_MODEL * D_INNER, 256), 256, 0, stream>>>(
            out_proj_w + (size_t)layer * D_MODEL * D_INNER, w_out_b, D_MODEL * D_INNER);

        // in_proj: (4096 x 3072) = act(4096x768) * w_in(3072x768)^T   (CK=256)
        gemm_bf16_nt<0><<<dim3(cdiv(2 * D_INNER, 128), M_ROWS / 64), 256, smem_bytes(256), stream>>>(
            act_b, w_in_b, xz, M_ROWS, 2 * D_INNER, D_MODEL, 256, nullptr);

        conv_silu<<<cdiv(EW, 256), 256, 0, stream>>>(
            xz, conv_w + (size_t)layer * D_INNER * D_CONV, conv_b + (size_t)layer * D_INNER, xs, xs_b);

        // x_proj: (4096 x 80) = xs(4096x1536) * w_xp(80x1536)^T   (CK=256)
        gemm_bf16_nt<0><<<dim3(cdiv(XP_OUT, 128), M_ROWS / 64), 256, smem_bytes(256), stream>>>(
            xs_b, w_xp_b, xdbl, M_ROWS, XP_OUT, D_INNER, 256, nullptr);

        // dt slice -> bf16, K padded 48 -> 64
        cvt_pad_bf16<<<cdiv(M_ROWS * DT_PAD, 256), 256, 0, stream>>>(
            xdbl, XP_OUT, DT_RANK, dt_b, DT_PAD, M_ROWS);

        // dt_proj + softplus(.. + bias): (4096 x 1536) = dt(4096x64) * w_dt(1536x64)^T (CK=64)
        gemm_bf16_nt<1><<<dim3(cdiv(D_INNER, 128), M_ROWS / 64), 256, smem_bytes(64), stream>>>(
            dt_b, w_dt_b, delta, M_ROWS, D_INNER, DT_PAD, 64, dt_proj_b + (size_t)layer * D_INNER);

        ssm_scan<<<cdiv(B_SZ * D_INNER, 256), 256, 0, stream>>>(
            delta, xs, xdbl, A_log + (size_t)layer * D_INNER * D_STATE, yssm);

        gate_silu<<<cdiv(EW, 256), 256, 0, stream>>>(
            yssm, xs, xz, Dvec + (size_t)layer * D_INNER, xs_b /* y_b */);

        // out_proj: (4096 x 768) = y(4096x1536) * w_out(768x1536)^T   (CK=256)
        gemm_bf16_nt<0><<<dim3(cdiv(D_MODEL, 128), M_ROWS / 64), 256, smem_bytes(256), stream>>>(
            xs_b, w_out_b, hout, M_ROWS, D_MODEL, D_INNER, 256, nullptr);
    }

    layernorm_row<<<M_ROWS, 256, 0, stream>>>(hout, ln_g, ln_b, (float*)d_out);
}
